// SerGINE_49160195670084
// MI455X (gfx1250) — matrix operand order repaired and checked by reading, not compiled, not run on hardware
//
#include <hip/hip_runtime.h>
#include <hip/hip_bf16.h>
#include <cstdint>

// Problem constants (match reference)
#define NA   200000
#define NF   50000
#define EA_  800000
#define EF_  200000
#define NG_  4096
#define DD   128
#define D2   256
#define BN_EPS 1e-5f

typedef __bf16 bf16_t;
typedef bf16_t v16bf __attribute__((ext_vector_type(16)));
typedef float  v8f   __attribute__((ext_vector_type(8)));
typedef float  f4v   __attribute__((ext_vector_type(4)));

__device__ __forceinline__ bf16_t f2bf(float f) {
    union { float f; uint32_t u; } v; v.f = f;
    uint32_t u = v.u;
    uint32_t r = u + 0x7FFFu + ((u >> 16) & 1u);   // round-to-nearest-even
    uint16_t h = (uint16_t)(r >> 16);
    union { uint16_t u; bf16_t b; } o; o.u = h;
    return o.b;
}

// ---------------------------------------------------------------------------
// Pack f32 weight matrix W[K,N] (row-major) into CDNA5 WMMA B-fragment order:
// frag[((kt*Ntiles + nt)*32 + lane)*16 + j] = bf16(W[(kt*32 + (lane>=16)*8 + (j<8 ? j : j+8))*N + nt*16 + lane%16])
// ---------------------------------------------------------------------------
__global__ void prep_wfrag_kernel(const float* __restrict__ W, bf16_t* __restrict__ out,
                                  int K, int N) {
    int idx = blockIdx.x * blockDim.x + threadIdx.x;
    int Nt = N >> 4;
    int total = (K >> 5) * Nt * 32;
    if (idx >= total) return;
    int lane = idx & 31;
    int tile = idx >> 5;
    int nt = tile % Nt;
    int kt = tile / Nt;
    int n  = (nt << 4) + (lane & 15);
    int k0 = (kt << 5) + ((lane >> 4) << 3);
    bf16_t* o = out + (size_t)idx * 16;
#pragma unroll
    for (int j = 0; j < 16; ++j) o[j] = f2bf(W[(size_t)(k0 + (j < 8 ? j : j + 8)) * N + n]);
}

// ---------------------------------------------------------------------------
// WMMA GEMM: C[M,N] = (A + Aadd)[M,K](f32) * Wfrag(bf16 prepacked) + bias[N]
// Aadd may be null (fuses the GINE "x + agg" residual into the A-fragment load).
// One wave -> 16 rows x 64 cols (4 accumulator tiles). blockDim = (32,4).
// ---------------------------------------------------------------------------
__global__ __launch_bounds__(128) void wmma_gemm_bias_kernel(
    const float* __restrict__ A, const float* __restrict__ Aadd,
    const bf16_t* __restrict__ Bf, const float* __restrict__ bias,
    float* __restrict__ C, int Mtiles, int K, int N) {
    int mtile = blockIdx.x * blockDim.y + threadIdx.y;
    if (mtile >= Mtiles) return;
    int lane = threadIdx.x;
    int half = lane >> 4;      // 0: K 0-7/16-23, 1: K 8-15/24-31
    int r    = lane & 15;      // A row within tile / C column within tile
    int Ntiles = N >> 4;
    int ntile0 = blockIdx.y << 2;     // 4 N-tiles per wave
    size_t rowoff = (size_t)(mtile * 16 + r) * K;
    const float* arow  = A + rowoff;
    const float* arow2 = Aadd ? (Aadd + rowoff) : nullptr;

    v8f acc[4] = {};
    int Kt = K >> 5;
    for (int kt = 0; kt < Kt; ++kt) {
        int kb = (kt << 5) + (half << 3);
        f4v a0 = *(const f4v*)(arow + kb);
        f4v a1 = *(const f4v*)(arow + kb + 4);
        f4v a2 = *(const f4v*)(arow + kb + 16);
        f4v a3 = *(const f4v*)(arow + kb + 20);
        if (arow2) {
            a0 = a0 + *(const f4v*)(arow2 + kb);
            a1 = a1 + *(const f4v*)(arow2 + kb + 4);
            a2 = a2 + *(const f4v*)(arow2 + kb + 16);
            a3 = a3 + *(const f4v*)(arow2 + kb + 20);
        }
        v16bf af;
        af[0]  = f2bf(a0[0]); af[1]  = f2bf(a0[1]); af[2]  = f2bf(a0[2]); af[3]  = f2bf(a0[3]);
        af[4]  = f2bf(a1[0]); af[5]  = f2bf(a1[1]); af[6]  = f2bf(a1[2]); af[7]  = f2bf(a1[3]);
        af[8]  = f2bf(a2[0]); af[9]  = f2bf(a2[1]); af[10] = f2bf(a2[2]); af[11] = f2bf(a2[3]);
        af[12] = f2bf(a3[0]); af[13] = f2bf(a3[1]); af[14] = f2bf(a3[2]); af[15] = f2bf(a3[3]);
#pragma unroll
        for (int nt = 0; nt < 4; ++nt) {
            const v16bf bmat =
                *(const v16bf*)(Bf + ((size_t)(kt * Ntiles + ntile0 + nt) * 32 + lane) * 16);
            acc[nt] = __builtin_amdgcn_wmma_f32_16x16x32_bf16(
                false, af, false, bmat, (short)0, acc[nt], false, false);
        }
    }
#pragma unroll
    for (int nt = 0; nt < 4; ++nt) {
        int col = (ntile0 + nt) * 16 + r;
        float bv = bias[col];
#pragma unroll
        for (int vr = 0; vr < 8; ++vr) {
            int row = mtile * 16 + vr + half * 8;
            C[(size_t)row * N + col] = acc[nt][vr] + bv;
        }
    }
}

// ---------------------------------------------------------------------------
// Row matvec for embeddings (K<=128, N==128): out = X @ W + b (+ extra)
// One block per row, 128 threads (one per output feature).
// ---------------------------------------------------------------------------
__global__ __launch_bounds__(128) void rowmv_kernel(
    const float* __restrict__ X, const float* __restrict__ W,
    const float* __restrict__ b, const float* __restrict__ extra,
    float* __restrict__ out, int K) {
    __shared__ float sx[128];
    int row = blockIdx.x;
    int t = threadIdx.x;
    if (t < K) sx[t] = X[(size_t)row * K + t];
    __syncthreads();
    float acc = b[t];
    for (int k = 0; k < K; ++k) acc += sx[k] * W[(size_t)k * DD + t];
    if (extra) acc += extra[(size_t)row * DD + t];
    out[(size_t)row * DD + t] = acc;
}

// ---------------------------------------------------------------------------
// Fused edge message + scatter-add (atom graph, edge-attr dim 11):
// agg[dst] += relu(h[src] + edge_attr@W + b), W cached in LDS, 8 edges/block.
// ---------------------------------------------------------------------------
__global__ __launch_bounds__(128) void atom_msg_kernel(
    const float* __restrict__ h, const float* __restrict__ ea,
    const int* __restrict__ src, const int* __restrict__ dst,
    const float* __restrict__ W, const float* __restrict__ b,
    float* __restrict__ agg, int E) {
    __shared__ float sW[11 * 128];
    __shared__ float sb[128];
    __shared__ float sea[8][12];
    int t = threadIdx.x;
    for (int i = t; i < 11 * 128; i += 128) sW[i] = W[i];
    sb[t] = b[t];
    int e0 = blockIdx.x * 8;
    for (int i = t; i < 8 * 11; i += 128) {
        int e = e0 + i / 11;
        if (e < E) sea[i / 11][i % 11] = ea[(size_t)e * 11 + (i % 11)];
    }
    __syncthreads();
    for (int k = 0; k < 8; ++k) {
        int e = e0 + k;
        if (e >= E) break;
        int s = src[e], d = dst[e];
        float acc = sb[t];
#pragma unroll
        for (int j = 0; j < 11; ++j) acc += sea[k][j] * sW[j * 128 + t];
        float m = h[(size_t)s * DD + t] + acc;
        m = m > 0.f ? m : 0.f;
        atomicAdd(&agg[(size_t)d * DD + t], m);
    }
}

// Same, FG graph (edge-attr dim 101): W is 101x128 = ~50.5 KB LDS, 4 edges/block.
__global__ __launch_bounds__(128) void fg_msg_kernel(
    const float* __restrict__ g, const float* __restrict__ ea,
    const int* __restrict__ src, const int* __restrict__ dst,
    const float* __restrict__ W, const float* __restrict__ b,
    float* __restrict__ agg, int E) {
    __shared__ float sW[101 * 128];
    __shared__ float sb[128];
    __shared__ float sea[4][104];
    int t = threadIdx.x;
    for (int i = t; i < 101 * 128; i += 128) sW[i] = W[i];
    sb[t] = b[t];
    int e0 = blockIdx.x * 4;
    for (int i = t; i < 4 * 101; i += 128) {
        int e = e0 + i / 101;
        if (e < E) sea[i / 101][i % 101] = ea[(size_t)e * 101 + (i % 101)];
    }
    __syncthreads();
    for (int k = 0; k < 4; ++k) {
        int e = e0 + k;
        if (e >= E) break;
        int s = src[e], d = dst[e];
        float acc = sb[t];
        for (int j = 0; j < 101; ++j) acc += sea[k][j] * sW[j * 128 + t];
        float m = g[(size_t)s * DD + t] + acc;
        m = m > 0.f ? m : 0.f;
        atomicAdd(&agg[(size_t)d * DD + t], m);
    }
}

// BN column statistics: blockDim == C, 512 rows per block, atomics into sum/sumsq.
__global__ void bn_stats_kernel(const float* __restrict__ X, float* __restrict__ sum,
                                float* __restrict__ sumsq, int M, int C) {
    int c = threadIdx.x;
    int r0 = blockIdx.x * 512;
    int r1 = r0 + 512; if (r1 > M) r1 = M;
    float s = 0.f, s2 = 0.f;
    for (int r = r0; r < r1; ++r) {
        float v = X[(size_t)r * C + c];
        s += v; s2 += v * v;
    }
    atomicAdd(&sum[c], s);
    atomicAdd(&sumsq[c], s2);
}

__global__ void bn_finalize_kernel(const float* __restrict__ sum, const float* __restrict__ sumsq,
                                   const float* __restrict__ gamma, const float* __restrict__ beta,
                                   float* __restrict__ scale, float* __restrict__ shift, int M) {
    int c = threadIdx.x;
    float inv = 1.f / (float)M;
    float mean = sum[c] * inv;
    float var  = sumsq[c] * inv - mean * mean;
    float sc   = gamma[c] * rsqrtf(var + BN_EPS);
    scale[c] = sc;
    shift[c] = beta[c] - mean * sc;
}

// Vectorized (float4) BN apply: y = x*scale[c] + shift[c], optional relu.
// n4 = n/4; C multiple of 4; pointers 16B aligned.
__global__ void bn_apply_kernel(float* __restrict__ X, const float* __restrict__ scale,
                                const float* __restrict__ shift, size_t n4, int C, int relu) {
    size_t i = (size_t)blockIdx.x * blockDim.x + threadIdx.x;
    if (i >= n4) return;
    size_t base = i * 4;
    int c = (int)(base % (size_t)C);
    f4v x  = *(const f4v*)(X + base);
    f4v sc = *(const f4v*)(scale + c);
    f4v sh = *(const f4v*)(shift + c);
    f4v y  = x * sc + sh;
    if (relu) {
#pragma unroll
        for (int j = 0; j < 4; ++j) y[j] = y[j] > 0.f ? y[j] : 0.f;
    }
    *(f4v*)(X + base) = y;
}

// a2f: sum[fg_idx[a]] += h[atom_idx[a]]; cnt[fg_idx[a]] += 1
__global__ void a2f_scatter_kernel(const float* __restrict__ h, const int* __restrict__ aidx,
                                   const int* __restrict__ fidx, float* __restrict__ sum,
                                   float* __restrict__ cnt) {
    size_t i = (size_t)blockIdx.x * blockDim.x + threadIdx.x;
    if (i >= (size_t)NA * DD) return;
    int a = (int)(i >> 7);
    int d = (int)(i & 127);
    int f = fidx[a];
    atomicAdd(&sum[(size_t)f * DD + d], h[(size_t)aidx[a] * DD + d]);
    if (d == 0) atomicAdd(&cnt[f], 1.f);
}

// final: sum[fg_batch[f]] += g[f]; cnt += 1
__global__ void seg_scatter_kernel(const float* __restrict__ g, const int* __restrict__ batch,
                                   float* __restrict__ sum, float* __restrict__ cnt) {
    size_t i = (size_t)blockIdx.x * blockDim.x + threadIdx.x;
    if (i >= (size_t)NF * DD) return;
    int f = (int)(i >> 7);
    int d = (int)(i & 127);
    int b = batch[f];
    atomicAdd(&sum[(size_t)b * DD + d], g[i]);
    if (d == 0) atomicAdd(&cnt[b], 1.f);
}

// Vectorized: out[i] = sum[i] / max(cnt[row], 1); n4 = n/4 (4 | DD)
__global__ void seg_div_kernel(const float* __restrict__ sum, const float* __restrict__ cnt,
                               float* __restrict__ out, size_t n4) {
    size_t i = (size_t)blockIdx.x * blockDim.x + threadIdx.x;
    if (i >= n4) return;
    size_t base = i * 4;
    int r = (int)(base >> 7);
    float c = cnt[r];
    c = c > 1.f ? c : 1.f;
    float rcp = 1.f / c;
    f4v s = *(const f4v*)(sum + base);
    f4v o = s * rcp;
    *(f4v*)(out + base) = o;
}

// ---------------------------------------------------------------------------
extern "C" void kernel_launch(void* const* d_in, const int* in_sizes, int n_in,
                              void* d_out, int out_size, void* d_ws, size_t ws_size,
                              hipStream_t stream) {
    (void)in_sizes; (void)n_in; (void)out_size; (void)ws_size;

    const float* atom_x      = (const float*)d_in[0];
    const float* fg_x        = (const float*)d_in[1];
    const float* atom_ea     = (const float*)d_in[2];
    const float* fg_ea       = (const float*)d_in[3];
    const float* atom_emb_W  = (const float*)d_in[4];
    const float* atom_emb_b  = (const float*)d_in[5];
    const float* fg_emb_W    = (const float*)d_in[6];
    const float* fg_emb_b    = (const float*)d_in[7];
    const float* bond_W      = (const float*)d_in[8];
    const float* bond_b      = (const float*)d_in[9];
    const float* ag_W1       = (const float*)d_in[10];
    const float* ag_b1       = (const float*)d_in[11];
    const float* ag_g1       = (const float*)d_in[12];
    const float* ag_be1      = (const float*)d_in[13];
    const float* ag_W2       = (const float*)d_in[14];
    const float* ag_b2       = (const float*)d_in[15];
    const float* abn_g       = (const float*)d_in[16];
    const float* abn_b       = (const float*)d_in[17];
    const float* fge_W       = (const float*)d_in[18];
    const float* fge_b       = (const float*)d_in[19];
    const float* fg_W1       = (const float*)d_in[20];
    const float* fg_b1       = (const float*)d_in[21];
    const float* fg_g1       = (const float*)d_in[22];
    const float* fg_be1      = (const float*)d_in[23];
    const float* fg_W2       = (const float*)d_in[24];
    const float* fg_b2       = (const float*)d_in[25];
    const float* fbn_g       = (const float*)d_in[26];
    const float* fbn_b       = (const float*)d_in[27];
    const float* a2f_W       = (const float*)d_in[28];
    const float* a2f_b       = (const float*)d_in[29];
    const int*   atom_ei     = (const int*)d_in[30];
    const int*   fg_ei       = (const int*)d_in[31];
    const int*   atom_idx    = (const int*)d_in[32];
    const int*   fg_idx      = (const int*)d_in[33];
    const int*   fg_batch    = (const int*)d_in[34];

    // ---- workspace layout ----
    char* wsb = (char*)d_ws;
    size_t off = 0;
    auto alloc = [&](size_t bytes) -> void* {
        off = (off + 255) & ~(size_t)255;
        void* p = wsb + off;
        off += bytes;
        return p;
    };
    float* h       = (float*)alloc((size_t)NA * DD * 4);
    float* agg     = (float*)alloc((size_t)NA * DD * 4);
    float* t2      = (float*)alloc((size_t)NA * D2 * 4);
    float* a2f_sum = (float*)alloc((size_t)NF * DD * 4);
    float* cnt     = (float*)alloc((size_t)NF * 4);       // NF >= NG
    float* outsum  = (float*)alloc((size_t)NG_ * DD * 4);
    float* bsum    = (float*)alloc(256 * 4);
    float* bsq     = (float*)alloc(256 * 4);
    float* bscale  = (float*)alloc(256 * 4);
    float* bshift  = (float*)alloc(256 * 4);
    bf16_t* fr_agW1 = (bf16_t*)alloc(3 * (size_t)DD * D2 * 2);
    bf16_t* fr_agW2 = (bf16_t*)alloc(3 * (size_t)D2 * DD * 2);
    bf16_t* fr_fgW1 = (bf16_t*)alloc(2 * (size_t)DD * D2 * 2);
    bf16_t* fr_fgW2 = (bf16_t*)alloc(2 * (size_t)D2 * DD * 2);
    bf16_t* fr_a2f  = (bf16_t*)alloc((size_t)DD * DD * 2);
    // FG-phase buffers alias the atom agg region (atom phase done by then)
    float* g       = agg;
    float* gagg    = agg + (size_t)NF * DD;
    float* a2f_out = agg + 2 * (size_t)NF * DD;
    float* gt2     = t2;

    auto prep = [&](const float* W, bf16_t* out, int K, int N) {
        int total = (K / 32) * (N / 16) * 32;
        prep_wfrag_kernel<<<(total + 255) / 256, 256, 0, stream>>>(W, out, K, N);
    };
    auto gemm = [&](const float* A, const float* Aadd, const bf16_t* Bf, const float* bias,
                    float* C, int M, int K, int N) {
        int Mtiles = M / 16;
        dim3 blk(32, 4);
        dim3 grd((unsigned)((Mtiles + 3) / 4), (unsigned)(N / 64));
        wmma_gemm_bias_kernel<<<grd, blk, 0, stream>>>(A, Aadd, Bf, bias, C, Mtiles, K, N);
    };
    auto bn = [&](float* X, int M, int C, const float* gam, const float* bet, int relu) {
        hipMemsetAsync(bsum, 0, 256 * 4, stream);
        hipMemsetAsync(bsq, 0, 256 * 4, stream);
        bn_stats_kernel<<<(M + 511) / 512, C, 0, stream>>>(X, bsum, bsq, M, C);
        bn_finalize_kernel<<<1, C, 0, stream>>>(bsum, bsq, gam, bet, bscale, bshift, M);
        size_t n4 = (size_t)M * C / 4;
        bn_apply_kernel<<<(unsigned)((n4 + 255) / 256), 256, 0, stream>>>(X, bscale, bshift, n4, C, relu);
    };

    // ---- pre-pack WMMA weight fragments (deterministic, every call) ----
    for (int i = 0; i < 3; ++i) {
        prep(ag_W1 + (size_t)i * DD * D2, fr_agW1 + (size_t)i * DD * D2, DD, D2);
        prep(ag_W2 + (size_t)i * D2 * DD, fr_agW2 + (size_t)i * D2 * DD, D2, DD);
    }
    for (int i = 0; i < 2; ++i) {
        prep(fg_W1 + (size_t)i * DD * D2, fr_fgW1 + (size_t)i * DD * D2, DD, D2);
        prep(fg_W2 + (size_t)i * D2 * DD, fr_fgW2 + (size_t)i * D2 * DD, D2, DD);
    }
    prep(a2f_W, fr_a2f, DD, DD);

    // ---- atom embedding ----
    rowmv_kernel<<<NA, 128, 0, stream>>>(atom_x, atom_emb_W, atom_emb_b, nullptr, h, 101);

    const int* asrc = atom_ei;
    const int* adst = atom_ei + EA_;
    size_t nAD = (size_t)NA * DD;

    // ---- 3 atom GINE layers ----
    for (int i = 0; i < 3; ++i) {
        hipMemsetAsync(agg, 0, nAD * 4, stream);
        atom_msg_kernel<<<(EA_ + 7) / 8, 128, 0, stream>>>(
            h, atom_ea, asrc, adst, bond_W + (size_t)i * 11 * DD, bond_b + (size_t)i * DD, agg, EA_);
        // A = h + agg fused into the WMMA A-fragment load
        gemm(h, agg, fr_agW1 + (size_t)i * DD * D2, ag_b1 + (size_t)i * D2, t2, NA, DD, D2);
        bn(t2, NA, D2, ag_g1 + (size_t)i * D2, ag_be1 + (size_t)i * D2, 1);
        gemm(t2, nullptr, fr_agW2 + (size_t)i * D2 * DD, ag_b2 + (size_t)i * DD, h, NA, D2, DD);
        bn(h, NA, DD, abn_g + (size_t)i * DD, abn_b + (size_t)i * DD, i != 2 ? 1 : 0);
    }

    // ---- atom -> FG pooling + projection ----
    hipMemsetAsync(a2f_sum, 0, (size_t)NF * DD * 4, stream);
    hipMemsetAsync(cnt, 0, (size_t)NF * 4, stream);
    a2f_scatter_kernel<<<(unsigned)((nAD + 255) / 256), 256, 0, stream>>>(h, atom_idx, fg_idx, a2f_sum, cnt);
    size_t nFD = (size_t)NF * DD;
    seg_div_kernel<<<(unsigned)((nFD / 4 + 255) / 256), 256, 0, stream>>>(a2f_sum, cnt, a2f_sum, nFD / 4);
    gemm(a2f_sum, nullptr, fr_a2f, a2f_b, a2f_out, NF, DD, DD);

    // ---- FG embedding (+ a2f) ----
    rowmv_kernel<<<NF, 128, 0, stream>>>(fg_x, fg_emb_W, fg_emb_b, a2f_out, g, 73);

    const int* fsrc = fg_ei;
    const int* fdst = fg_ei + EF_;

    // ---- 2 FG GINE layers ----
    for (int i = 0; i < 2; ++i) {
        hipMemsetAsync(gagg, 0, nFD * 4, stream);
        fg_msg_kernel<<<(EF_ + 3) / 4, 128, 0, stream>>>(
            g, fg_ea, fsrc, fdst, fge_W + (size_t)i * 101 * DD, fge_b + (size_t)i * DD, gagg, EF_);
        gemm(g, gagg, fr_fgW1 + (size_t)i * DD * D2, fg_b1 + (size_t)i * D2, gt2, NF, DD, D2);
        bn(gt2, NF, D2, fg_g1 + (size_t)i * D2, fg_be1 + (size_t)i * D2, 1);
        gemm(gt2, nullptr, fr_fgW2 + (size_t)i * D2 * DD, fg_b2 + (size_t)i * DD, g, NF, D2, DD);
        bn(g, NF, DD, fbn_g + (size_t)i * DD, fbn_b + (size_t)i * DD, i != 1 ? 1 : 0);
    }

    // ---- final per-graph mean ----
    hipMemsetAsync(outsum, 0, (size_t)NG_ * DD * 4, stream);
    hipMemsetAsync(cnt, 0, (size_t)NG_ * 4, stream);
    seg_scatter_kernel<<<(unsigned)((nFD + 255) / 256), 256, 0, stream>>>(g, fg_batch, outsum, cnt);
    size_t nOut4 = (size_t)NG_ * DD / 4;
    seg_div_kernel<<<(unsigned)((nOut4 + 255) / 256), 256, 0, stream>>>(outsum, cnt, (float*)d_out, nOut4);
}
